// NeuralLongTermMemory_21157008900313
// MI455X (gfx1250) — matrix operand order repaired and checked by reading, not compiled, not run on hardware
//
#include <hip/hip_runtime.h>
#include <cstdint>
#include <cstddef>

// Problem constants (match reference)
#define BB 4
#define SS 8192
#define DD 512
#define HH 1024
#define KW 4
#define BS (BB * SS)
#define MEM_LR 0.01f
#define MEM_MOM 0.9f
#define EPSV 1e-12f

typedef __attribute__((ext_vector_type(16))) __bf16 v16bf;
typedef __attribute__((ext_vector_type(8)))  float  v8f;
typedef unsigned int u32x4 __attribute__((ext_vector_type(4)));
typedef int          i32x8 __attribute__((ext_vector_type(8)));
typedef int          i32x4 __attribute__((ext_vector_type(4)));

__device__ __forceinline__ unsigned short f2bf(float f) {
  unsigned int u = __float_as_uint(f);
  u += 0x7fffu + ((u >> 16) & 1u);          // round-to-nearest-even
  return (unsigned short)(u >> 16);
}
__device__ __forceinline__ float sigm(float v) { return 1.f / (1.f + expf(-v)); }
__device__ __forceinline__ float silu(float v) { return v * sigm(v); }

enum { EPI_F32 = 0, EPI_BF16 = 1, EPI_SILU_BF16 = 2, EPI_SILU_BOTH = 3 };

// ---------------------------------------------------------------------------
// TDM: DMA one 2-D bf16 tile (tile_w x tile_h elements, row stride in elems)
// from global memory into LDS at lds_byte_off, padding each row so the LDS
// row stride matches the compute layout (pad codes per D# group1 field spec).
// D# packing per CDNA5 ISA 8.3/8.4. Issued per-wave; tracked by TENSORcnt.
// This toolchain carries the 6-arg builtin form (extra int32x8 group).
// ---------------------------------------------------------------------------
__device__ __forceinline__ void tdm_load_2d_bf16(uint32_t lds_byte_off,
                                                 const unsigned short* gsrc,
                                                 uint32_t stride_elems,
                                                 uint32_t tile_w_elems,
                                                 uint32_t tile_h,
                                                 uint32_t pad_interval_code,
                                                 uint32_t pad_amount_code)
{
  const uint64_t ga = (uint64_t)(uintptr_t)gsrc;
  u32x4 g0;
  g0[0] = 1u;                                   // count=1 valid user descriptor
  g0[1] = lds_byte_off;                         // lds_addr
  g0[2] = (uint32_t)ga;                         // global_addr[31:0]
  g0[3] = (uint32_t)((ga >> 32) & 0x01ffffffu)  // global_addr[56:32]
          | 0x80000000u;                        // type=2 ("image")
  // group1: mask=0 | data_size=2B | pad_enable | pad_interval | pad_amount
  const uint32_t td0 = 0x00100000u;             // oversized tensor dims:
  const uint32_t td1 = 0x00100000u;             // interior tiles never clip
  i32x8 g1;
  g1[0] = (int)((1u << 16) | (1u << 20) |
                (pad_interval_code << 22) | (pad_amount_code << 25));
  g1[1] = (int)((td0 & 0xffffu) << 16);                       // abar=0 | td0.lo
  g1[2] = (int)((td0 >> 16) | ((td1 & 0xffffu) << 16));       // td0.hi | td1.lo
  g1[3] = (int)((td1 >> 16) | (tile_w_elems << 16));          // td1.hi | tile_dim0
  g1[4] = (int)(tile_h & 0xffffu);                            // tile_dim1 | tile_dim2=0
  g1[5] = (int)stride_elems;                                  // dim0_stride[31:0]
  g1[6] = 0;                                                  // stride.hi | dim1_stride.lo
  g1[7] = 0;
  const i32x4 z4 = {0, 0, 0, 0};                // 2-D tile: groups 2/3 unused
  const i32x8 z8 = {0, 0, 0, 0, 0, 0, 0, 0};
  __builtin_amdgcn_tensor_load_to_lds(g0, g1, z4, z4, z8, 0);
}

// ---------------------------------------------------------------------------
// Tiled bf16 WMMA GEMM: C(MxN) = A(MxK) * B^T-or-B, fp32 accumulate.
//  ATRANS: A stored in memory as (K,M)  (used for weight-gradient GEMMs)
//  BNK   : B stored as (N,K) torch-Linear weight (y = x @ W^T); else (K,N)
// Block: 256 threads = 8 wave32; tile 128x128, K-step 32.
// Non-transposed tiles are staged via TDM (tensor_load_to_lds + tensorcnt);
// transposed tiles are staged manually (TDM has no transpose).
// ---------------------------------------------------------------------------
template<bool ATRANS, bool BNK, int EPI>
__global__ __launch_bounds__(256)
void gemm_wmma_bf16(const unsigned short* __restrict__ A,
                    const unsigned short* __restrict__ Bm,
                    float* __restrict__ Cf,
                    unsigned short* __restrict__ Cb,
                    int M, int N, int K)
{
  constexpr int TM = 128, TN = 128, TK = 32, LS = 40; // LS: lds row stride (bf16)
  constexpr bool TDMA = !ATRANS;   // direct A tile -> TDM
  constexpr bool TDMB = BNK;       // direct B tile -> TDM
  __shared__ unsigned short sa[TM * LS];
  __shared__ unsigned short sb[TN * LS];

  const int ntn = N / TN;
  const int m0 = (blockIdx.x / ntn) * TM;
  const int n0 = (blockIdx.x % ntn) * TN;
  const int t   = threadIdx.x;
  const int wv  = t >> 5;       // wave 0..7
  const int ln  = t & 31;
  const int hf  = ln >> 4;      // lane half
  const int l16 = ln & 15;

  const uint32_t saOff = (uint32_t)(uintptr_t)(&sa[0]);
  const uint32_t sbOff = (uint32_t)(uintptr_t)(&sb[0]);

  v8f acc[8] = {};

  for (int k0 = 0; k0 < K; k0 += TK) {
    // ---- stage A tile into LDS as [m][k], row stride LS=40 bf16 (80B) ----
    if (TDMA) {
      if (wv == 0) {
        // rows of 32 bf16 = 16 dwords (code 3), pad 4 dwords (code 3) -> 80B
        tdm_load_2d_bf16(saOff, A + (size_t)m0 * K + k0, (uint32_t)K, TK, TM, 3u, 3u);
      }
    } else {
      const int k = t >> 3, ms = (t & 7) * 16;
      const float4* src = (const float4*)(A + (size_t)(k0 + k) * M + m0 + ms);
      float4 v0 = src[0], v1 = src[1];
      unsigned short tmp[16];
      *(float4*)(tmp) = v0; *(float4*)(tmp + 8) = v1;
      #pragma unroll
      for (int j = 0; j < 16; j++) sa[(ms + j) * LS + k] = tmp[j];
    }
    // ---- stage B tile into LDS as [n][k] ----
    if (TDMB) {
      if (wv == 0) {
        tdm_load_2d_bf16(sbOff, Bm + (size_t)n0 * K + k0, (uint32_t)K, TK, TN, 3u, 3u);
      }
    } else {
      const int k = t >> 3, ns = (t & 7) * 16;
      const float4* src = (const float4*)(Bm + (size_t)(k0 + k) * N + n0 + ns);
      float4 v0 = src[0], v1 = src[1];
      unsigned short tmp[16];
      *(float4*)(tmp) = v0; *(float4*)(tmp + 8) = v1;
      #pragma unroll
      for (int j = 0; j < 16; j++) sb[(ns + j) * LS + k] = tmp[j];
    }
    if (TDMA || TDMB) {
      if (wv == 0) __builtin_amdgcn_s_wait_tensorcnt(0);
    }
    __syncthreads();

    // A fragment (ISA 7.12.2 16-bit A layout: VGPR v -> K pair {2v|2v+8}+8*half)
    union { unsigned int u[8]; v16bf v; } af;
    {
      const unsigned short* ar = sa + (wv * 16 + l16) * LS;
      #pragma unroll
      for (int i = 0; i < 8; i++) {
        const int kk = ((i < 4) ? 2 * i : 2 * i + 8) + 8 * hf;
        af.u[i] = *(const unsigned int*)(ar + kk);
      }
    }
    #pragma unroll
    for (int nt = 0; nt < 8; nt++) {
      union { unsigned int u[8]; v16bf v; } bfr;
      const unsigned short* br = sb + (nt * 16 + l16) * LS;
      #pragma unroll
      for (int i = 0; i < 8; i++) {
        const int kk = ((i < 4) ? 2 * i : 2 * i + 8) + 8 * hf;   // B K-striping: same as A (K {2v|2v+8}+8*half)
        bfr.u[i] = *(const unsigned int*)(br + kk);
      }
      acc[nt] = __builtin_amdgcn_wmma_f32_16x16x32_bf16(
          false, af.v, false, bfr.v, (short)0, acc[nt], false, false);
    }
    __syncthreads();
  }

  // epilogue: C/D layout VGPR i -> M = i + 8*half, N = lane16
  #pragma unroll
  for (int nt = 0; nt < 8; nt++) {
    const int n = n0 + nt * 16 + l16;
    #pragma unroll
    for (int i = 0; i < 8; i++) {
      const int m = m0 + wv * 16 + i + 8 * hf;
      const size_t idx = (size_t)m * N + n;
      const float val = acc[nt][i];
      if (EPI == EPI_F32)            { Cf[idx] = val; }
      else if (EPI == EPI_BF16)      { Cb[idx] = f2bf(val); }
      else if (EPI == EPI_SILU_BF16) { Cb[idx] = f2bf(silu(val)); }
      else                           { Cf[idx] = val; Cb[idx] = f2bf(silu(val)); }
    }
  }
}

// ---------------------------------------------------------------------------
// Causal depthwise conv (K=4, cross-correlation) + SiLU, optional row L2-norm.
// One block per (b,s) token, 512 threads = channels.
// ---------------------------------------------------------------------------
template<bool L2N, bool OUTBF>
__global__ __launch_bounds__(512)
void conv_post_kernel(const float* __restrict__ in, const float* __restrict__ w,
                      const float* __restrict__ bias,
                      float* __restrict__ outf, unsigned short* __restrict__ outb)
{
  const int bs = blockIdx.x;
  const int b = bs / SS, s = bs % SS;
  const int d = threadIdx.x;
  float acc = bias[d];
  #pragma unroll
  for (int j = 0; j < KW; j++) {
    const int sj = s - (KW - 1) + j;
    if (sj >= 0) acc += in[((size_t)b * SS + sj) * DD + d] * w[d * KW + j];
  }
  float y = silu(acc);
  if (L2N) {
    __shared__ float red[DD];
    red[d] = y * y;
    __syncthreads();
    for (int st = DD / 2; st > 0; st >>= 1) {
      if (d < st) red[d] += red[d + st];
      __syncthreads();
    }
    const float nrm = sqrtf(red[0]);
    y /= fmaxf(nrm, EPSV);
  }
  if (OUTBF) outb[(size_t)bs * DD + d] = f2bf(y);
  else       outf[(size_t)bs * DD + d] = y;
}

__global__ void xmean_kernel(const float* __restrict__ x, float* __restrict__ xm)
{
  const int idx = blockIdx.x * blockDim.x + threadIdx.x;
  if (idx >= BB * DD) return;
  const int b = idx / DD, d = idx % DD;
  const float* p = x + (size_t)b * SS * DD + d;
  float s = 0.f;
  for (int t = 0; t < SS; t++) s += p[(size_t)t * DD];
  xm[idx] = s / (float)SS;
}

// Three scalar gates from x_mean: alpha, theta(=*LR), eta(=*MOM)
__global__ __launch_bounds__(512)
void gates_kernel(const float* __restrict__ xm,
                  const float* __restrict__ wd, const float* __restrict__ bd,
                  const float* __restrict__ wl, const float* __restrict__ bl,
                  const float* __restrict__ wm, const float* __restrict__ bm,
                  float* __restrict__ gout)
{
  const int i = threadIdx.x;                       // output unit 0..511
  const float* Ws[3] = {wd, wl, wm};
  const float* Bs[3] = {bd, bl, bm};
  float sums[3];
  #pragma unroll
  for (int g = 0; g < 3; g++) {
    const float* Wr = Ws[g] + (size_t)i * DD;
    float sg = 0.f;
    for (int b = 0; b < BB; b++) {
      float a = Bs[g][i];
      const float* xr = xm + b * DD;
      for (int j = 0; j < DD; j++) a += xr[j] * Wr[j];
      sg += sigm(a);
    }
    sums[g] = sg;
  }
  __shared__ float red[DD];
  const float scl[3] = {1.f, MEM_LR, MEM_MOM};
  for (int g = 0; g < 3; g++) {
    red[i] = sums[g];
    __syncthreads();
    for (int st = DD / 2; st > 0; st >>= 1) {
      if (i < st) red[i] += red[i + st];
      __syncthreads();
    }
    if (i == 0) gout[g] = (red[0] / (float)(BB * DD)) * scl[g];
    __syncthreads();
  }
}

__global__ void cvt_bf16_kernel(const float* __restrict__ in,
                                unsigned short* __restrict__ out, size_t n)
{
  for (size_t i = blockIdx.x * (size_t)blockDim.x + threadIdx.x; i < n;
       i += (size_t)gridDim.x * blockDim.x)
    out[i] = f2bf(in[i]);
}

__global__ void dpred_kernel(const float* __restrict__ pred, const float* __restrict__ vals,
                             unsigned short* __restrict__ out, int n)
{
  const float scale = 2.f / (float)((size_t)BS * DD);
  for (int i = blockIdx.x * blockDim.x + threadIdx.x; i < n; i += gridDim.x * blockDim.x)
    out[i] = f2bf((pred[i] - vals[i]) * scale);
}

__global__ void dh1_kernel(const float* __restrict__ da1, const float* __restrict__ h1,
                           unsigned short* __restrict__ out, int n)
{
  for (int i = blockIdx.x * blockDim.x + threadIdx.x; i < n; i += gridDim.x * blockDim.x) {
    const float h = h1[i], sg = sigm(h);
    out[i] = f2bf(da1[i] * (sg * (1.f + h * (1.f - sg))));
  }
}

__global__ void update_kernel(const float* __restrict__ w, const float* __restrict__ mom,
                              const float* __restrict__ g, float* __restrict__ nw,
                              float* __restrict__ sout, const float* __restrict__ gates, int n)
{
  const float alpha = gates[0], theta = gates[1], eta = gates[2];
  for (int i = blockIdx.x * blockDim.x + threadIdx.x; i < n; i += gridDim.x * blockDim.x) {
    const float sv = eta * mom[i] - theta * g[i];
    sout[i] = sv;
    nw[i] = (1.f - alpha) * w[i] + sv;
  }
}

// ---------------------------------------------------------------------------
extern "C" void kernel_launch(void* const* d_in, const int* in_sizes, int n_in,
                              void* d_out, int out_size, void* d_ws, size_t ws_size,
                              hipStream_t stream)
{
  (void)in_sizes; (void)n_in; (void)out_size; (void)ws_size;
  const float* x        = (const float*)d_in[0];
  const float* w_k1     = (const float*)d_in[1];
  const float* w_k2     = (const float*)d_in[2];
  const float* w_v1     = (const float*)d_in[3];
  const float* w_v2     = (const float*)d_in[4];
  const float* w_q1     = (const float*)d_in[5];
  const float* w_q2     = (const float*)d_in[6];
  const float* conv_k_w = (const float*)d_in[7];
  const float* conv_k_b = (const float*)d_in[8];
  const float* conv_v_w = (const float*)d_in[9];
  const float* conv_v_b = (const float*)d_in[10];
  const float* conv_q_w = (const float*)d_in[11];
  const float* conv_q_b = (const float*)d_in[12];
  const float* gate_d_w = (const float*)d_in[13];
  const float* gate_d_b = (const float*)d_in[14];
  const float* gate_l_w = (const float*)d_in[15];
  const float* gate_l_b = (const float*)d_in[16];
  const float* gate_m_w = (const float*)d_in[17];
  const float* gate_m_b = (const float*)d_in[18];
  const float* w_out    = (const float*)d_in[19];
  const float* mem_w1   = (const float*)d_in[20];
  const float* mem_w2   = (const float*)d_in[21];
  const float* mom1     = (const float*)d_in[22];
  const float* mom2     = (const float*)d_in[23];

  float* out = (float*)d_out;
  float* out_y  = out;                                   // (B,S,D)
  float* out_n1 = out + (size_t)BS * DD;                 // (H,D)
  float* out_n2 = out_n1 + (size_t)HH * DD;              // (D,H)
  float* out_s1 = out_n2 + (size_t)DD * HH;
  float* out_s2 = out_s1 + (size_t)HH * DD;

  // scratch carve-out
  uint8_t* wsb = (uint8_t*)d_ws;
  size_t off = 0;
  auto alloc = [&](size_t bytes) -> void* {
    void* p = wsb + off; off += (bytes + 255) & ~(size_t)255; return p;
  };
  unsigned short* xb   = (unsigned short*)alloc((size_t)BS * DD * 2);
  unsigned short* wk1b = (unsigned short*)alloc((size_t)DD * DD * 2);
  unsigned short* wk2b = (unsigned short*)alloc((size_t)DD * DD * 2);
  unsigned short* wv1b = (unsigned short*)alloc((size_t)DD * DD * 2);
  unsigned short* wv2b = (unsigned short*)alloc((size_t)DD * DD * 2);
  unsigned short* wq1b = (unsigned short*)alloc((size_t)DD * DD * 2);
  unsigned short* wq2b = (unsigned short*)alloc((size_t)DD * DD * 2);
  unsigned short* wob  = (unsigned short*)alloc((size_t)DD * DD * 2);
  unsigned short* wm1b = (unsigned short*)alloc((size_t)HH * DD * 2);
  unsigned short* wm2b = (unsigned short*)alloc((size_t)DD * HH * 2);
  unsigned short* hidb = (unsigned short*)alloc((size_t)BS * HH * 2); // proj hid / retr hid / a1
  unsigned short* kb   = (unsigned short*)alloc((size_t)BS * DD * 2); // keys bf16
  unsigned short* qb   = (unsigned short*)alloc((size_t)BS * DD * 2); // queries bf16
  unsigned short* rb   = (unsigned short*)alloc((size_t)BS * DD * 2); // retrieved bf16
  unsigned short* dpb  = (unsigned short*)alloc((size_t)BS * DD * 2); // dpred bf16
  unsigned short* dhb  = (unsigned short*)alloc((size_t)BS * HH * 2); // dh1 bf16
  float* projf = (float*)alloc((size_t)BS * DD * 4);  // proj scratch, later pred
  float* vf    = (float*)alloc((size_t)BS * DD * 4);  // values fp32
  float* h1f   = (float*)alloc((size_t)BS * HH * 4);  // h1 fp32
  float* da1f  = (float*)alloc((size_t)BS * HH * 4);  // da1 fp32
  float* xmf   = (float*)alloc((size_t)BB * DD * 4);
  float* gbuf  = (float*)alloc(64);
  float* g1f   = (float*)alloc((size_t)HH * DD * 4);
  float* g2f   = (float*)alloc((size_t)DD * HH * 4);

  const dim3 b256(256);
  auto gg = [](int M, int N) { return dim3((unsigned)((M / 128) * (N / 128))); };

  // ---- fp32 -> bf16 staging ----
  cvt_bf16_kernel<<<2048, b256, 0, stream>>>(x, xb, (size_t)BS * DD);
  cvt_bf16_kernel<<<256, b256, 0, stream>>>(w_k1, wk1b, (size_t)DD * DD);
  cvt_bf16_kernel<<<256, b256, 0, stream>>>(w_k2, wk2b, (size_t)DD * DD);
  cvt_bf16_kernel<<<256, b256, 0, stream>>>(w_v1, wv1b, (size_t)DD * DD);
  cvt_bf16_kernel<<<256, b256, 0, stream>>>(w_v2, wv2b, (size_t)DD * DD);
  cvt_bf16_kernel<<<256, b256, 0, stream>>>(w_q1, wq1b, (size_t)DD * DD);
  cvt_bf16_kernel<<<256, b256, 0, stream>>>(w_q2, wq2b, (size_t)DD * DD);
  cvt_bf16_kernel<<<256, b256, 0, stream>>>(w_out, wob, (size_t)DD * DD);
  cvt_bf16_kernel<<<512, b256, 0, stream>>>(mem_w1, wm1b, (size_t)HH * DD);
  cvt_bf16_kernel<<<512, b256, 0, stream>>>(mem_w2, wm2b, (size_t)DD * HH);

  // ---- K/V/Q projections: silu(x@W1^T)@W2^T, then causal conv + post ----
  gemm_wmma_bf16<false, true, EPI_SILU_BF16><<<gg(BS, DD), b256, 0, stream>>>(xb, wk1b, nullptr, hidb, BS, DD, DD);
  gemm_wmma_bf16<false, true, EPI_F32><<<gg(BS, DD), b256, 0, stream>>>(hidb, wk2b, projf, nullptr, BS, DD, DD);
  conv_post_kernel<true, true><<<BS, 512, 0, stream>>>(projf, conv_k_w, conv_k_b, nullptr, kb);

  gemm_wmma_bf16<false, true, EPI_SILU_BF16><<<gg(BS, DD), b256, 0, stream>>>(xb, wv1b, nullptr, hidb, BS, DD, DD);
  gemm_wmma_bf16<false, true, EPI_F32><<<gg(BS, DD), b256, 0, stream>>>(hidb, wv2b, projf, nullptr, BS, DD, DD);
  conv_post_kernel<false, false><<<BS, 512, 0, stream>>>(projf, conv_v_w, conv_v_b, vf, nullptr);

  gemm_wmma_bf16<false, true, EPI_SILU_BF16><<<gg(BS, DD), b256, 0, stream>>>(xb, wq1b, nullptr, hidb, BS, DD, DD);
  gemm_wmma_bf16<false, true, EPI_F32><<<gg(BS, DD), b256, 0, stream>>>(hidb, wq2b, projf, nullptr, BS, DD, DD);
  conv_post_kernel<true, true><<<BS, 512, 0, stream>>>(projf, conv_q_w, conv_q_b, nullptr, qb);

  // ---- gates (device-side scalars; graph-capture safe) ----
  xmean_kernel<<<(BB * DD + 255) / 256, b256, 0, stream>>>(x, xmf);
  gates_kernel<<<1, 512, 0, stream>>>(xmf, gate_d_w, gate_d_b, gate_l_w, gate_l_b,
                                      gate_m_w, gate_m_b, gbuf);

  // ---- retrieval + output projection ----
  gemm_wmma_bf16<false, true, EPI_SILU_BF16><<<gg(BS, HH), b256, 0, stream>>>(qb, wm1b, nullptr, hidb, BS, HH, DD);
  gemm_wmma_bf16<false, true, EPI_BF16><<<gg(BS, DD), b256, 0, stream>>>(hidb, wm2b, nullptr, rb, BS, DD, HH);
  gemm_wmma_bf16<false, true, EPI_F32><<<gg(BS, DD), b256, 0, stream>>>(rb, wob, out_y, nullptr, BS, DD, DD);

  // ---- memory-MLP gradient ----
  // h1 = keys@W1^T (fp32) and a1 = silu(h1) (bf16) in one pass
  gemm_wmma_bf16<false, true, EPI_SILU_BOTH><<<gg(BS, HH), b256, 0, stream>>>(kb, wm1b, h1f, hidb, BS, HH, DD);
  // pred = a1 @ W2^T
  gemm_wmma_bf16<false, true, EPI_F32><<<gg(BS, DD), b256, 0, stream>>>(hidb, wm2b, projf, nullptr, BS, DD, HH);
  dpred_kernel<<<2048, b256, 0, stream>>>(projf, vf, dpb, BS * DD);
  // da1 = dpred @ W2   (B stored (K,N))
  gemm_wmma_bf16<false, false, EPI_F32><<<gg(BS, HH), b256, 0, stream>>>(dpb, wm2b, da1f, nullptr, BS, HH, DD);
  dh1_kernel<<<2048, b256, 0, stream>>>(da1f, h1f, dhb, BS * HH);
  // g2 = dpred^T @ a1   (M=D, N=H, K=BS)
  gemm_wmma_bf16<true, false, EPI_F32><<<gg(DD, HH), b256, 0, stream>>>(dpb, hidb, g2f, nullptr, DD, HH, BS);
  // g1 = dh1^T @ keys   (M=H, N=D, K=BS)
  gemm_wmma_bf16<true, false, EPI_F32><<<gg(HH, DD), b256, 0, stream>>>(dhb, kb, g1f, nullptr, HH, DD, BS);

  // ---- momentum + decay updates -> d_out ----
  update_kernel<<<512, b256, 0, stream>>>(mem_w1, mom1, g1f, out_n1, out_s1, gbuf, HH * DD);
  update_kernel<<<512, b256, 0, stream>>>(mem_w2, mom2, g2f, out_n2, out_s2, gbuf, DD * HH);
}
